// TransformerBlock_2104533975442
// MI455X (gfx1250) — matrix operand order repaired and checked by reading, not compiled, not run on hardware
//
#include <hip/hip_runtime.h>
#include <hip/hip_bf16.h>
#include <cstdint>
#include <cstddef>

// ---------------- types ----------------
typedef _Float16 h8  __attribute__((ext_vector_type(8)));   // 16 bytes
typedef _Float16 h16 __attribute__((ext_vector_type(16)));  // 32 bytes (8 VGPRs)
typedef float    f8  __attribute__((ext_vector_type(8)));   // 32 bytes (8 VGPRs)
typedef int      i4  __attribute__((ext_vector_type(4)));   // 16 bytes

union F16Frag { h16 v; h8 h[2]; };

__device__ __forceinline__ f8 wmma_f16(h16 a, h16 b, f8 c) {
  // D = A(16x32 f16) * B(32x16 f16) + C(16x16 f32)
  return __builtin_amdgcn_wmma_f32_16x16x32_f16(false, a, false, b, (short)0, c,
                                                false, false);
}

// ---------------- async global->LDS copy (CDNA5) ----------------
#define AS1 __attribute__((address_space(1)))
#define AS3 __attribute__((address_space(3)))

#if __has_builtin(__builtin_amdgcn_global_load_async_to_lds_b128) && \
    __has_builtin(__builtin_amdgcn_s_wait_asynccnt)
#define HAS_ASYNC_LDS 1
#else
#define HAS_ASYNC_LDS 0
#endif

// copy 16 bytes (8 f16) global -> LDS
__device__ __forceinline__ void cp16_g2l(_Float16* l, const _Float16* g) {
#if HAS_ASYNC_LDS
  _Float16* gm = const_cast<_Float16*>(g);
  __builtin_amdgcn_global_load_async_to_lds_b128((AS1 i4*)gm, (AS3 i4*)l, 0, 0);
#else
  *(h8*)l = *(const h8*)g;
#endif
}

__device__ __forceinline__ void wait_async_lds() {
#if HAS_ASYNC_LDS
  __builtin_amdgcn_s_wait_asynccnt(0);
#endif
}

// ---------------- constants ----------------
#define DIMC   2048
#define NHEAD  16
#define HDIM   128
#define HIDC   5632
#define BC     2
#define SC     2048

// ================= fp32 -> f16 convert =================
__global__ __launch_bounds__(256) void k_cvt_f16(const float* __restrict__ in,
                                                 _Float16* __restrict__ out,
                                                 size_t n) {
  size_t i = (size_t)blockIdx.x * blockDim.x + threadIdx.x;
  size_t stride = (size_t)gridDim.x * blockDim.x;
  for (; i < n; i += stride) out[i] = (_Float16)in[i];
}

// ================= RMSNorm (f32 in, f16 out) =================
__global__ __launch_bounds__(256) void k_rmsnorm(const float* __restrict__ X,
                                                 const float* __restrict__ W,
                                                 _Float16* __restrict__ Y,
                                                 int ncols) {
  const int row = blockIdx.x;
  const float* x = X + (size_t)row * ncols;
  float ss = 0.f;
  for (int i = threadIdx.x; i < ncols; i += 256) { float v = x[i]; ss += v * v; }
#pragma unroll
  for (int off = 16; off >= 1; off >>= 1) ss += __shfl_xor(ss, off, 32);
  __shared__ float red[8];
  if ((threadIdx.x & 31) == 0) red[threadIdx.x >> 5] = ss;
  __syncthreads();
  float tot = 0.f;
#pragma unroll
  for (int i = 0; i < 8; ++i) tot += red[i];
  const float scale = rsqrtf(tot / (float)ncols + 1e-5f);
  _Float16* y = Y + (size_t)row * ncols;
  for (int i = threadIdx.x; i < ncols; i += 256)
    y[i] = (_Float16)(x[i] * scale * W[i]);
}

// ================= WMMA GEMM: C[M,N] = A[M,K] @ B[N,K]^T (+resid) ============
// A,B row-major f16 (K contiguous).  Block tile 128x128x64, 8 waves,
// each wave: 64x32 via 4x2 accum tiles, 16 v_wmma per K-step.
// Double-buffered LDS filled with GLOBAL_LOAD_ASYNC_TO_LDS_B128 (ASYNCcnt).
#define BM 128
#define BN 128
#define BK 64
#define LDT 72  // f16 stride (64 + 8 pad), 144B: 16B-aligned rows

template <bool RES, bool F32OUT, bool F16OUT>
__global__ __launch_bounds__(256) void k_gemm_f16(
    const _Float16* __restrict__ A, const _Float16* __restrict__ Bm,
    int M, int N, int K,
    const float* __restrict__ resid,
    float* __restrict__ Cf, _Float16* __restrict__ Ch) {
  __shared__ _Float16 As[2][BM * LDT];
  __shared__ _Float16 Bs[2][BN * LDT];

  const int tid  = threadIdx.x;
  const int wave = tid >> 5;
  const int lane = tid & 31;
  const int lh   = lane >> 4;   // 0/1 : K-half selector
  const int ll   = lane & 15;   // row/col within tile

  const int m0 = blockIdx.y * BM;
  const int n0 = blockIdx.x * BN;
  const int wm = (wave >> 2) * 64;  // wave M-offset in block tile
  const int wn = (wave & 3) * 32;   // wave N-offset in block tile

  const int lrow = tid >> 1;          // 0..127 (tile row for staging)
  const int lseg = (tid & 1) * 32;    // 0 / 32 (f16 offset: half a 64-wide row)

  f8 acc[4][2];
#pragma unroll
  for (int i = 0; i < 4; ++i)
#pragma unroll
    for (int j = 0; j < 2; ++j) { f8 z = {}; acc[i][j] = z; }

  // stage one 128x64 A tile + 128x64 B tile into buffer `buf`
  auto stage = [&](int buf, int kb) {
    const _Float16* ga = A + (size_t)(m0 + lrow) * K + kb + lseg;
    const _Float16* gb = Bm + (size_t)(n0 + lrow) * K + kb + lseg;
    _Float16* la = &As[buf][lrow * LDT + lseg];
    _Float16* lb = &Bs[buf][lrow * LDT + lseg];
#pragma unroll
    for (int s = 0; s < 4; ++s) {       // 4 x 16B per matrix per thread
      cp16_g2l(la + s * 8, ga + s * 8);
      cp16_g2l(lb + s * 8, gb + s * 8);
    }
  };

  stage(0, 0);
  int buf = 0;

  for (int kb = 0; kb < K; kb += BK) {
    wait_async_lds();
    __syncthreads();
    if (kb + BK < K) stage(buf ^ 1, kb + BK);  // overlap next-tile copy

#pragma unroll
    for (int kk = 0; kk < BK; kk += 32) {
      F16Frag afr[4], bfr[2];
#pragma unroll
      for (int i = 0; i < 4; ++i) {
        const int r = wm + i * 16 + ll;  // A-frag: lane ll = row M
        afr[i].h[0] = *(const h8*)&As[buf][r * LDT + kk + lh * 8];
        afr[i].h[1] = *(const h8*)&As[buf][r * LDT + kk + lh * 8 + 16];
      }
#pragma unroll
      for (int j = 0; j < 2; ++j) {
        const int c = wn + j * 16 + ll;  // B-frag: lane ll = col N
        bfr[j].h[0] = *(const h8*)&Bs[buf][c * LDT + kk + lh * 8];
        bfr[j].h[1] = *(const h8*)&Bs[buf][c * LDT + kk + lh * 8 + 16];
      }
#pragma unroll
      for (int i = 0; i < 4; ++i)
#pragma unroll
        for (int j = 0; j < 2; ++j)
          acc[i][j] = wmma_f16(afr[i].v, bfr[j].v, acc[i][j]);
    }
    __syncthreads();
    buf ^= 1;
  }

  // ---- epilogue: C layout VGPR r -> row = r + 8*lh, lane ll -> col
#pragma unroll
  for (int i = 0; i < 4; ++i)
#pragma unroll
    for (int j = 0; j < 2; ++j) {
      const int row0 = m0 + wm + i * 16 + lh * 8;
      const int col  = n0 + wn + j * 16 + ll;
#pragma unroll
      for (int r = 0; r < 8; ++r) {
        const size_t idx = (size_t)(row0 + r) * N + col;
        float val = acc[i][j][r];
        if constexpr (RES) val += resid[idx];
        if constexpr (F32OUT) Cf[idx] = val;
        if constexpr (F16OUT) Ch[idx] = (_Float16)val;
      }
    }
}

// ================= Flash attention =================
// grid (S/64, NHEAD, B), block 128 (4 waves, 16 q-rows each).
// q/k/v/o layout: [B, S, NHEAD, HDIM] f16.  bias [1,H,S,S], mask [1,1,S,S] f32.
#define QT 64
#define KT 32
#define QLD 136  // 128 + 8 pad
#define VLD 40   // 32 + 8 pad

__global__ __launch_bounds__(128) void k_flash(
    const _Float16* __restrict__ Q, const _Float16* __restrict__ Km,
    const _Float16* __restrict__ Vm, const float* __restrict__ bias,
    const float* __restrict__ mask, _Float16* __restrict__ O,
    int S_, float scale) {
  __shared__ _Float16 Qs[QT * QLD];
  __shared__ _Float16 Ks[KT * QLD];
  __shared__ _Float16 Vts[HDIM * VLD];  // transposed: [d][key]
  __shared__ _Float16 Ps[QT * VLD];     // probs f16: [qrow][key]

  const int tid  = threadIdx.x;
  const int wave = tid >> 5;
  const int lane = tid & 31;
  const int lh   = lane >> 4;
  const int ll   = lane & 15;

  const int b  = blockIdx.z;
  const int h  = blockIdx.y;
  const int q0 = blockIdx.x * QT;

  auto gidx = [&](int s, int d) -> size_t {
    return (((size_t)b * S_ + s) * NHEAD + h) * HDIM + d;
  };

  // ---- load Q tile 64x128 via async copies (512 segs of 16 f16)
#pragma unroll
  for (int it = 0; it < 4; ++it) {
    const int seg = tid + it * 128;
    const int r = seg >> 3, d = (seg & 7) * 16;
    cp16_g2l(&Qs[r * QLD + d],     &Q[gidx(q0 + r, d)]);
    cp16_g2l(&Qs[r * QLD + d + 8], &Q[gidx(q0 + r, d) + 8]);
  }

  f8 acc[8];
#pragma unroll
  for (int t = 0; t < 8; ++t) { f8 z = {}; acc[t] = z; }
  float m_[8], l_[8];
#pragma unroll
  for (int r = 0; r < 8; ++r) { m_[r] = -1e30f; l_[r] = 0.f; }

  for (int kt = 0; kt < S_; kt += KT) {
    // ---- K tile 32x128 (async)
#pragma unroll
    for (int it = 0; it < 2; ++it) {
      const int seg = tid + it * 128;
      const int r = seg >> 3, d = (seg & 7) * 16;
      cp16_g2l(&Ks[r * QLD + d],     &Km[gidx(kt + r, d)]);
      cp16_g2l(&Ks[r * QLD + d + 8], &Km[gidx(kt + r, d) + 8]);
    }
    // ---- V tile 32x128 -> transposed Vts[d][key] (manual, needs relayout)
#pragma unroll
    for (int it = 0; it < 4; ++it) {
      const int seg = tid + it * 128;
      const int key = seg >> 4, d0 = (seg & 15) * 8;
      h8 vv = *(const h8*)&Vm[gidx(kt + key, d0)];
#pragma unroll
      for (int j = 0; j < 8; ++j) Vts[(d0 + j) * VLD + key] = vv[j];
    }
    wait_async_lds();
    __syncthreads();

    // ---- scores 16x32 per wave: Q(16x128) @ K^T(128x32)
    f8 sc[2];
    { f8 z = {}; sc[0] = z; sc[1] = z; }
#pragma unroll
    for (int ds = 0; ds < HDIM; ds += 32) {
      F16Frag qf;
      const int qr = wave * 16 + ll;
      qf.h[0] = *(const h8*)&Qs[qr * QLD + ds + lh * 8];
      qf.h[1] = *(const h8*)&Qs[qr * QLD + ds + lh * 8 + 16];
#pragma unroll
      for (int j = 0; j < 2; ++j) {
        F16Frag kf;
        const int kc = j * 16 + ll;
        kf.h[0] = *(const h8*)&Ks[kc * QLD + ds + lh * 8];
        kf.h[1] = *(const h8*)&Ks[kc * QLD + ds + lh * 8 + 16];
        sc[j] = wmma_f16(qf.v, kf.v, sc[j]);
      }
    }

    // ---- scale + bias + mask, online softmax
#pragma unroll
    for (int r = 0; r < 8; ++r) {
      const int qrow = q0 + wave * 16 + r + 8 * lh;
#pragma unroll
      for (int j = 0; j < 2; ++j) {
        const int col = kt + j * 16 + ll;
        const size_t bo = ((size_t)h * S_ + qrow) * S_ + col;
        sc[j][r] = sc[j][r] * scale + bias[bo] + mask[(size_t)qrow * S_ + col];
      }
    }
#pragma unroll
    for (int r = 0; r < 8; ++r) {
      float mx = fmaxf(sc[0][r], sc[1][r]);
#pragma unroll
      for (int off = 8; off >= 1; off >>= 1) mx = fmaxf(mx, __shfl_xor(mx, off, 32));
      mx = fmaxf(mx, m_[r]);
      const float corr = __expf(m_[r] - mx);
      const float p0 = __expf(sc[0][r] - mx);
      const float p1 = __expf(sc[1][r] - mx);
      sc[0][r] = p0; sc[1][r] = p1;
      float sum = p0 + p1;
#pragma unroll
      for (int off = 8; off >= 1; off >>= 1) sum += __shfl_xor(sum, off, 32);
      l_[r] = l_[r] * corr + sum;
      m_[r] = mx;
#pragma unroll
      for (int t = 0; t < 8; ++t) acc[t][r] = acc[t][r] * corr;
    }

    // ---- P -> LDS (C-layout -> A-layout relayout), then P @ V
#pragma unroll
    for (int r = 0; r < 8; ++r) {
      const int pr = wave * 16 + r + 8 * lh;
      Ps[pr * VLD + ll]      = (_Float16)sc[0][r];
      Ps[pr * VLD + 16 + ll] = (_Float16)sc[1][r];
    }
    F16Frag pf;
    {
      const int pr = wave * 16 + ll;
      pf.h[0] = *(const h8*)&Ps[pr * VLD + lh * 8];
      pf.h[1] = *(const h8*)&Ps[pr * VLD + lh * 8 + 16];
    }
#pragma unroll
    for (int t = 0; t < 8; ++t) {
      F16Frag vf;
      const int d = t * 16 + ll;
      vf.h[0] = *(const h8*)&Vts[d * VLD + lh * 8];
      vf.h[1] = *(const h8*)&Vts[d * VLD + lh * 8 + 16];
      acc[t] = wmma_f16(pf.v, vf.v, acc[t]);
    }
    __syncthreads();
  }

  // ---- epilogue: O = acc / l
#pragma unroll
  for (int t = 0; t < 8; ++t) {
#pragma unroll
    for (int r = 0; r < 8; ++r) {
      const int qrow = q0 + wave * 16 + r + 8 * lh;
      const int d = t * 16 + ll;
      O[gidx(qrow, d)] = (_Float16)(acc[t][r] / l_[r]);
    }
  }
}

// ================= SwiGLU elementwise =================
__global__ __launch_bounds__(256) void k_swiglu(const _Float16* __restrict__ a1,
                                                const _Float16* __restrict__ a3,
                                                _Float16* __restrict__ g,
                                                size_t n) {
  size_t i = (size_t)blockIdx.x * blockDim.x + threadIdx.x;
  size_t stride = (size_t)gridDim.x * blockDim.x;
  for (; i < n; i += stride) {
    float x = (float)a1[i];
    float s = x / (1.f + __expf(-x));
    g[i] = (_Float16)(s * (float)a3[i]);
  }
}

// ================= host orchestration =================
extern "C" void kernel_launch(void* const* d_in, const int* in_sizes, int n_in,
                              void* d_out, int out_size, void* d_ws, size_t ws_size,
                              hipStream_t stream) {
  (void)in_sizes; (void)n_in; (void)out_size; (void)ws_size;

  const float* x         = (const float*)d_in[0];
  const float* mask      = (const float*)d_in[1];
  const float* attn_bias = (const float*)d_in[2];
  const float* wq        = (const float*)d_in[3];
  const float* wk        = (const float*)d_in[4];
  const float* wv        = (const float*)d_in[5];
  const float* wo        = (const float*)d_in[6];
  const float* w1        = (const float*)d_in[7];
  const float* w2        = (const float*)d_in[8];
  const float* w3        = (const float*)d_in[9];
  const float* anw       = (const float*)d_in[10];
  const float* fnw       = (const float*)d_in[11];
  float* out = (float*)d_out;

  const size_t BS = (size_t)BC * SC;          // 4096 tokens
  const size_t szDD = (size_t)DIMC * DIMC;    // 2048*2048
  const size_t szHD = (size_t)HIDC * DIMC;    // 5632*2048
  const size_t szTD = BS * DIMC;              // 4096*2048
  const size_t szTH = BS * HIDC;              // 4096*5632

  char* base = (char*)d_ws;
  size_t off = 0;
  auto alloc = [&](size_t bytes) -> void* {
    void* p = base + off;
    off += (bytes + 255) & ~(size_t)255;
    return p;
  };

  _Float16* wq16 = (_Float16*)alloc(szDD * 2);
  _Float16* wk16 = (_Float16*)alloc(szDD * 2);
  _Float16* wv16 = (_Float16*)alloc(szDD * 2);
  _Float16* wo16 = (_Float16*)alloc(szDD * 2);
  _Float16* w1h  = (_Float16*)alloc(szHD * 2);
  _Float16* w3h  = (_Float16*)alloc(szHD * 2);
  _Float16* w2h  = (_Float16*)alloc(szHD * 2);
  _Float16* hf   = (_Float16*)alloc(szTD * 2);
  _Float16* q16  = (_Float16*)alloc(szTD * 2);
  _Float16* k16  = (_Float16*)alloc(szTD * 2);
  _Float16* v16  = (_Float16*)alloc(szTD * 2);
  _Float16* o16  = (_Float16*)alloc(szTD * 2);
  float*    h1   = (float*)alloc(szTD * 4);
  _Float16* fn16 = (_Float16*)alloc(szTD * 2);
  _Float16* a1h  = (_Float16*)alloc(szTH * 2);
  _Float16* a3h  = (_Float16*)alloc(szTH * 2);
  _Float16* g16  = (_Float16*)alloc(szTH * 2);

  // ---- weight conversion
  k_cvt_f16<<<2048, 256, 0, stream>>>(wq, wq16, szDD);
  k_cvt_f16<<<2048, 256, 0, stream>>>(wk, wk16, szDD);
  k_cvt_f16<<<2048, 256, 0, stream>>>(wv, wv16, szDD);
  k_cvt_f16<<<2048, 256, 0, stream>>>(wo, wo16, szDD);
  k_cvt_f16<<<2048, 256, 0, stream>>>(w1, w1h, szHD);
  k_cvt_f16<<<2048, 256, 0, stream>>>(w3, w3h, szHD);
  k_cvt_f16<<<2048, 256, 0, stream>>>(w2, w2h, szHD);

  // ---- attention path
  k_rmsnorm<<<(int)BS, 256, 0, stream>>>(x, anw, hf, DIMC);

  dim3 gProj(DIMC / BN, (int)BS / BM);
  k_gemm_f16<false, false, true><<<gProj, 256, 0, stream>>>(
      hf, wq16, (int)BS, DIMC, DIMC, nullptr, nullptr, q16);
  k_gemm_f16<false, false, true><<<gProj, 256, 0, stream>>>(
      hf, wk16, (int)BS, DIMC, DIMC, nullptr, nullptr, k16);
  k_gemm_f16<false, false, true><<<gProj, 256, 0, stream>>>(
      hf, wv16, (int)BS, DIMC, DIMC, nullptr, nullptr, v16);

  dim3 gAttn(SC / QT, NHEAD, BC);
  k_flash<<<gAttn, 128, 0, stream>>>(q16, k16, v16, attn_bias, mask, o16, SC,
                                     0.0883883476483184f /* 1/sqrt(128) */);

  // h1 = x + attn_out @ wo^T
  k_gemm_f16<true, true, false><<<gProj, 256, 0, stream>>>(
      o16, wo16, (int)BS, DIMC, DIMC, x, h1, nullptr);

  // ---- FFN path
  k_rmsnorm<<<(int)BS, 256, 0, stream>>>(h1, fnw, fn16, DIMC);

  dim3 gUp(HIDC / BN, (int)BS / BM);
  k_gemm_f16<false, false, true><<<gUp, 256, 0, stream>>>(
      fn16, w1h, (int)BS, HIDC, DIMC, nullptr, nullptr, a1h);
  k_gemm_f16<false, false, true><<<gUp, 256, 0, stream>>>(
      fn16, w3h, (int)BS, HIDC, DIMC, nullptr, nullptr, a3h);

  k_swiglu<<<2048, 256, 0, stream>>>(a1h, a3h, g16, szTH);

  // out = h1 + g @ w2^T
  dim3 gDown(DIMC / BN, (int)BS / BM);
  k_gemm_f16<true, true, false><<<gDown, 256, 0, stream>>>(
      g16, w2h, (int)BS, DIMC, HIDC, h1, out, nullptr);
}
